// MultiHeadAttention_62801011802793
// MI455X (gfx1250) — matrix operand order repaired and checked by reading, not compiled, not run on hardware
//
#include <hip/hip_runtime.h>
#include <hip/hip_bf16.h>

// Problem constants
#define NB   2
#define NT   2048
#define ND   1024
#define NNH  16
#define NHD  64
#define MTOT (NB*NT)   // 4096 rows

typedef __attribute__((ext_vector_type(16))) __bf16 bf16x16;
typedef __attribute__((ext_vector_type(8)))  __bf16 bf16x8;
typedef __attribute__((ext_vector_type(8)))  float  f32x8;
typedef __attribute__((ext_vector_type(4)))  int    v4i;

#define AS1 __attribute__((address_space(1)))
#define AS3 __attribute__((address_space(3)))

typedef AS1 v4i* gptr_b128;   // global-AS pointer to 16B block
typedef AS3 v4i* lptr_b128;   // LDS-AS pointer to 16B block

#if defined(__has_builtin)
#if __has_builtin(__builtin_amdgcn_global_load_async_to_lds_b128)
#define HAVE_ASYNC_LDS 1
#endif
#endif
#ifndef HAVE_ASYNC_LDS
#define HAVE_ASYNC_LDS 0
#endif

union Frag16 { bf16x16 v; bf16x8 h[2]; };

__device__ __forceinline__ unsigned short f2bf(float f) {
  union { float f; unsigned u; } c; c.f = f;
  unsigned u = c.u;
  u += 0x7fffu + ((u >> 16) & 1u);   // round-to-nearest-even
  return (unsigned short)(u >> 16);
}

// Copy 16 bytes global -> LDS. Async (ASYNCcnt-tracked, VGPR-bypassing) when
// the toolchain exposes the gfx1250 builtin; plain VGPR bounce otherwise.
__device__ __forceinline__ void g2l_b128(unsigned short* ldsDst,
                                         const unsigned short* gSrc) {
#if HAVE_ASYNC_LDS
  // flat-shared addresses carry the LDS offset in the low 32 bits (ISA 10.2);
  // cast through integers to get exactly-typed AS1/AS3 v4i pointers.
  gptr_b128 g = (gptr_b128)(unsigned long long)gSrc;
  lptr_b128 l = (lptr_b128)(unsigned)(unsigned long long)ldsDst;
  __builtin_amdgcn_global_load_async_to_lds_b128(g, l, 0, 0);
#else
  *(uint4*)ldsDst = *(const uint4*)gSrc;
#endif
}

__device__ __forceinline__ void wait_async0() {
#if HAVE_ASYNC_LDS
#if defined(__has_builtin) && __has_builtin(__builtin_amdgcn_s_wait_asynccnt)
  __builtin_amdgcn_s_wait_asynccnt(0);
#else
  asm volatile("s_wait_asynccnt 0" ::: "memory");
#endif
#endif
}

// ---------------------------------------------------------------- fp32 -> bf16
__global__ void cvt_f32_bf16(const float* __restrict__ in,
                             unsigned short* __restrict__ out, int n) {
  int i = blockIdx.x * blockDim.x + threadIdx.x;
  if (i < n) out[i] = f2bf(in[i]);
}

// fp32 [K x 1024] -> bf16 transposed [N x 1024] (weights: W^T, K-contiguous)
__global__ void cvt_f32_bf16_T(const float* __restrict__ in,
                               unsigned short* __restrict__ out) {
  int i = blockIdx.x * blockDim.x + threadIdx.x;
  if (i < ND * ND) {
    int k = i >> 10, n = i & (ND - 1);
    out[(size_t)n * ND + k] = f2bf(in[i]);
  }
}

// ---------------------------------------------------------------- bf16 GEMM
// C[M x 1024] = A[M x 1024] * W[1024 x 1024], W given pre-transposed (WT[n][k]).
// Block tile 128x128, K-step 32, double-buffered async LDS staging.
// 8 waves (wave32) in 4x2 grid; each wave computes 32x64 = 2x4 WMMA tiles.
// MODE 0: bf16 out scattered to [B][NH][T][HD]   (Q, K projections)
// MODE 2: bf16 out scattered to [B][NH][HD][T]   (V projection, pre-transposed)
// MODE 1: fp32 out + bias to [M x 1024]          (output projection)
template<int MODE>
__global__ __launch_bounds__(256) void gemm_bf16(
    const unsigned short* __restrict__ A,
    const unsigned short* __restrict__ WT,
    unsigned short* __restrict__ outBf,
    float* __restrict__ outF,
    const float* __restrict__ bias)
{
  __shared__ __align__(16) unsigned short ldsA[2][128][40];  // [m][k], pad 8
  __shared__ __align__(16) unsigned short ldsB[2][128][40];  // [n][k], pad 8

  const int m0 = blockIdx.x * 128;
  const int n0 = blockIdx.y * 128;
  const int tid  = threadIdx.x;
  const int wave = tid >> 5, lane = tid & 31;
  const int hf   = lane >> 4, lm = lane & 15;
  const int wm   = wave >> 1, wn = wave & 1;

  // staging map: thread -> (row, 16-element segment); both A and WT rows are
  // K-contiguous so every copy is a b128 pair.
  const int srow = tid >> 1, sseg = (tid & 1) * 16;
  const unsigned short* gA = A  + (size_t)(m0 + srow) * ND + sseg;
  const unsigned short* gB = WT + (size_t)(n0 + srow) * ND + sseg;

  f32x8 acc[2][4] = {};

  // prologue: stage k0 = 0 into buffer 0
  g2l_b128(&ldsA[0][srow][sseg],     gA);
  g2l_b128(&ldsA[0][srow][sseg + 8], gA + 8);
  g2l_b128(&ldsB[0][srow][sseg],     gB);
  g2l_b128(&ldsB[0][srow][sseg + 8], gB + 8);
  wait_async0();
  __syncthreads();

  for (int k0 = 0; k0 < ND; k0 += 32) {
    const int cur = (k0 >> 5) & 1;
    if (k0 + 32 < ND) {  // prefetch next K-step into the other buffer
      const int nxt = cur ^ 1;
      g2l_b128(&ldsA[nxt][srow][sseg],     gA + k0 + 32);
      g2l_b128(&ldsA[nxt][srow][sseg + 8], gA + k0 + 40);
      g2l_b128(&ldsB[nxt][srow][sseg],     gB + k0 + 32);
      g2l_b128(&ldsB[nxt][srow][sseg + 8], gB + k0 + 40);
    }

    Frag16 a[2], b[4];
#pragma unroll
    for (int mi = 0; mi < 2; ++mi) {          // A 16x32: lane half -> K runs of 8
      int r = wm * 32 + mi * 16 + lm;
      a[mi].h[0] = *(const bf16x8*)&ldsA[cur][r][hf * 8];
      a[mi].h[1] = *(const bf16x8*)&ldsA[cur][r][16 + hf * 8];
    }
#pragma unroll
    for (int ni = 0; ni < 4; ++ni) {          // B 32x16: lane half -> K runs of 8
      int c = wn * 64 + ni * 16 + lm;
      b[ni].h[0] = *(const bf16x8*)&ldsB[cur][c][hf * 8];
      b[ni].h[1] = *(const bf16x8*)&ldsB[cur][c][16 + hf * 8];
    }
#pragma unroll
    for (int mi = 0; mi < 2; ++mi)
#pragma unroll
      for (int ni = 0; ni < 4; ++ni)
        acc[mi][ni] = __builtin_amdgcn_wmma_f32_16x16x32_bf16(
            false, a[mi].v, false, b[ni].v, (short)0, acc[mi][ni], false, false);

    wait_async0();
    __syncthreads();
  }

  // Epilogue. C layout: comp i, lane -> (m = i + 8*hf, n = lm)
#pragma unroll
  for (int mi = 0; mi < 2; ++mi)
#pragma unroll
    for (int ni = 0; ni < 4; ++ni)
#pragma unroll
      for (int i = 0; i < 8; ++i) {
        int gm = m0 + wm * 32 + mi * 16 + i + 8 * hf;
        int gn = n0 + wn * 64 + ni * 16 + lm;
        float v = acc[mi][ni][i];
        if (MODE == 0) {
          int bb = gm >> 11, t = gm & (NT - 1);
          int h = gn >> 6, hd = gn & 63;
          outBf[(((size_t)(bb * NNH + h)) * NT + t) * NHD + hd] = f2bf(v);
        } else if (MODE == 2) {
          int bb = gm >> 11, t = gm & (NT - 1);
          int h = gn >> 6, hd = gn & 63;
          outBf[(((size_t)(bb * NNH + h)) * NHD + hd) * NT + t] = f2bf(v);
        } else {
          outF[(size_t)gm * ND + gn] = v + bias[gn];
        }
      }
}

// ---------------------------------------------------------------- attention
// Grid: (B*NH, T/128). 8 waves; wave w owns query rows [qb+16w, qb+16w+16).
// Flash-attention with online softmax; all matmuls via WMMA bf16.
// V arrives pre-transposed ([B][NH][HD][T]) so all staging is contiguous b128.
__global__ __launch_bounds__(256) void attn_fwd(
    const unsigned short* __restrict__ Q,
    const unsigned short* __restrict__ K,
    const unsigned short* __restrict__ Vt,
    unsigned short* __restrict__ ctx)
{
  __shared__ __align__(16) unsigned short ldsK[64][72];       // [key][hd]
  __shared__ __align__(16) unsigned short ldsVT[64][72];      // [hd][key]
  __shared__ __align__(16) unsigned short ldsP[8][16][72];    // per-wave P scratch

  const int bh = blockIdx.x;
  const int qb = blockIdx.y * 128;
  const int tid  = threadIdx.x;
  const int wave = tid >> 5, lane = tid & 31;
  const int hf   = lane >> 4, lm = lane & 15;

  const unsigned short* Qp = Q  + (size_t)bh * NT * NHD;
  const unsigned short* Kp = K  + (size_t)bh * NT * NHD;
  const unsigned short* Vp = Vt + (size_t)bh * NHD * NT;

  // Q A-fragments (16 rows x 64 hd = two 16x32 k-steps), straight from global
  Frag16 qf[2];
  {
    int qr = qb + wave * 16 + lm;
    const unsigned short* p = Qp + (size_t)qr * NHD;
#pragma unroll
    for (int ks = 0; ks < 2; ++ks) {
      qf[ks].h[0] = *(const bf16x8*)(p + ks * 32 + hf * 8);
      qf[ks].h[1] = *(const bf16x8*)(p + ks * 32 + 16 + hf * 8);
    }
  }

  float m_run[8], l_run[8];
#pragma unroll
  for (int i = 0; i < 8; ++i) { m_run[i] = -3.0e38f; l_run[i] = 0.f; }
  f32x8 o[4] = {};

  // staging map: rowS in [0,64), 16-element segment
  const int rowS = tid >> 2, segS = (tid & 3) * 16;

  const int ntiles = (qb + 128) >> 6;   // key tiles of 64, causal upper bound
  for (int kt = 0; kt < ntiles; ++kt) {
    const int kb = kt * 64;
    __syncthreads();
    // stage K (key-major) and V^T (hd-major): all contiguous b128 copies
    g2l_b128(&ldsK[rowS][segS],      Kp + (size_t)(kb + rowS) * NHD + segS);
    g2l_b128(&ldsK[rowS][segS + 8],  Kp + (size_t)(kb + rowS) * NHD + segS + 8);
    g2l_b128(&ldsVT[rowS][segS],     Vp + (size_t)rowS * NT + kb + segS);
    g2l_b128(&ldsVT[rowS][segS + 8], Vp + (size_t)rowS * NT + kb + segS + 8);
    wait_async0();
    __syncthreads();

    // S = Q @ K^T  (16 x 64), contraction over hd
    f32x8 s[4] = {};
#pragma unroll
    for (int ks = 0; ks < 2; ++ks) {
#pragma unroll
      for (int ni = 0; ni < 4; ++ni) {
        Frag16 kf;
        int c = ni * 16 + lm;                 // N = key
        kf.h[0] = *(const bf16x8*)&ldsK[c][ks * 32 + hf * 8];
        kf.h[1] = *(const bf16x8*)&ldsK[c][ks * 32 + 16 + hf * 8];
        s[ni] = __builtin_amdgcn_wmma_f32_16x16x32_bf16(
            false, qf[ks].v, false, kf.v, (short)0, s[ni], false, false);
      }
    }

    // causal mask + scale (1/sqrt(64)) + online softmax
#pragma unroll
    for (int i = 0; i < 8; ++i) {
      const int qg = qb + wave * 16 + i + 8 * hf;
      float mx = -3.0e38f;
#pragma unroll
      for (int ni = 0; ni < 4; ++ni) {
        int kg = kb + ni * 16 + lm;
        float sv = (kg > qg) ? -3.0e38f : s[ni][i] * 0.125f;
        s[ni][i] = sv;
        mx = fmaxf(mx, sv);
      }
#pragma unroll
      for (int off = 1; off < 16; off <<= 1)   // reduce across the 16-lane row group
        mx = fmaxf(mx, __shfl_xor(mx, off, 32));
      float mnew  = fmaxf(m_run[i], mx);
      float alpha = __expf(m_run[i] - mnew);
      float rsum = 0.f;
#pragma unroll
      for (int ni = 0; ni < 4; ++ni) {
        float p = __expf(s[ni][i] - mnew);
        s[ni][i] = p;
        rsum += p;
      }
#pragma unroll
      for (int off = 1; off < 16; off <<= 1)
        rsum += __shfl_xor(rsum, off, 32);
      l_run[i] = l_run[i] * alpha + rsum;
      m_run[i] = mnew;
#pragma unroll
      for (int oi = 0; oi < 4; ++oi) o[oi][i] *= alpha;
    }

    // Re-layout P (C-frag) -> A-frag via per-wave LDS
#pragma unroll
    for (int ni = 0; ni < 4; ++ni)
#pragma unroll
      for (int i = 0; i < 8; ++i)
        ldsP[wave][i + 8 * hf][ni * 16 + lm] = f2bf(s[ni][i]);
    asm volatile("s_wait_dscnt 0" ::: "memory");  // intra-wave LDS RAW ordering

    // O += P @ V  (contraction over keys)
#pragma unroll
    for (int ks = 0; ks < 2; ++ks) {
      Frag16 pf;
      pf.h[0] = *(const bf16x8*)&ldsP[wave][lm][ks * 32 + hf * 8];
      pf.h[1] = *(const bf16x8*)&ldsP[wave][lm][ks * 32 + 16 + hf * 8];
#pragma unroll
      for (int oi = 0; oi < 4; ++oi) {
        Frag16 vf;
        int c = oi * 16 + lm;                 // N = hd
        vf.h[0] = *(const bf16x8*)&ldsVT[c][ks * 32 + hf * 8];
        vf.h[1] = *(const bf16x8*)&ldsVT[c][ks * 32 + 16 + hf * 8];
        o[oi] = __builtin_amdgcn_wmma_f32_16x16x32_bf16(
            false, pf.v, false, vf.v, (short)0, o[oi], false, false);
      }
    }
  }

  // finalize: ctx[b][t][h*64+hd], bf16, row-major [B*T, D]
  const int bb = bh / NNH, h = bh % NNH;
#pragma unroll
  for (int i = 0; i < 8; ++i) {
    float inv = 1.0f / l_run[i];
    int qg = qb + wave * 16 + i + 8 * hf;
    size_t rowoff = ((size_t)(bb * NT + qg)) * ND + h * NHD;
#pragma unroll
    for (int oi = 0; oi < 4; ++oi)
      ctx[rowoff + oi * 16 + lm] = f2bf(o[oi][i] * inv);
  }
}

// ---------------------------------------------------------------- launch
extern "C" void kernel_launch(void* const* d_in, const int* in_sizes, int n_in,
                              void* d_out, int out_size, void* d_ws, size_t ws_size,
                              hipStream_t stream) {
  (void)in_sizes; (void)n_in; (void)out_size; (void)ws_size;

  const float* x  = (const float*)d_in[0];
  const float* Wq = (const float*)d_in[1];
  const float* Wk = (const float*)d_in[2];
  const float* Wv = (const float*)d_in[3];
  const float* Wo = (const float*)d_in[4];
  const float* bo = (const float*)d_in[5];
  float* out = (float*)d_out;

  // Workspace layout (40 MB total): xb reused as ctx after attention.
  char* ws = (char*)d_ws;
  const size_t MB = 1024 * 1024;
  unsigned short* xb  = (unsigned short*)(ws);             // 8 MB
  unsigned short* wqT = (unsigned short*)(ws + 8  * MB);   // 2 MB each, W^T
  unsigned short* wkT = (unsigned short*)(ws + 10 * MB);
  unsigned short* wvT = (unsigned short*)(ws + 12 * MB);
  unsigned short* woT = (unsigned short*)(ws + 14 * MB);
  unsigned short* Qb  = (unsigned short*)(ws + 16 * MB);   // 8 MB each
  unsigned short* Kb  = (unsigned short*)(ws + 24 * MB);
  unsigned short* Vt  = (unsigned short*)(ws + 32 * MB);   // [B][NH][HD][T]
  unsigned short* ctx = xb;

  const int nx = MTOT * ND;   // 4M
  const int nw = ND * ND;     // 1M
  cvt_f32_bf16 <<<(nx + 255) / 256, 256, 0, stream>>>(x,  xb,  nx);
  cvt_f32_bf16_T<<<(nw + 255) / 256, 256, 0, stream>>>(Wq, wqT);
  cvt_f32_bf16_T<<<(nw + 255) / 256, 256, 0, stream>>>(Wk, wkT);
  cvt_f32_bf16_T<<<(nw + 255) / 256, 256, 0, stream>>>(Wv, wvT);
  cvt_f32_bf16_T<<<(nw + 255) / 256, 256, 0, stream>>>(Wo, woT);

  dim3 gg(MTOT / 128, ND / 128);       // 32 x 8
  gemm_bf16<0><<<gg, 256, 0, stream>>>(xb, wqT, Qb, nullptr, nullptr);
  gemm_bf16<0><<<gg, 256, 0, stream>>>(xb, wkT, Kb, nullptr, nullptr);
  gemm_bf16<2><<<gg, 256, 0, stream>>>(xb, wvT, Vt, nullptr, nullptr);

  dim3 ga(NB * NNH, NT / 128);         // 32 x 16
  attn_fwd<<<ga, 256, 0, stream>>>(Qb, Kb, Vt, ctx);

  gemm_bf16<1><<<gg, 256, 0, stream>>>(ctx, woT, nullptr, out, bo);
}
